// GNResidule_75050258530717
// MI455X (gfx1250) — matrix operand order repaired and checked by reading, not compiled, not run on hardware
//
#include <hip/hip_runtime.h>
#include <hip/hip_bf16.h>
#include <type_traits>

// ---------------------------------------------------------------------------
// GN network on MI455X (gfx1250), wave32 + v_wmma_f32_16x16x32_bf16.
//
// Activations/weights live as split bf16 planes (hi, lo); each GEMM does
// A@W ~= Ah@Wh + Ah@Wl + Al@Wh (~fp32 accuracy). Inner loop is pure
// saddr-form global loads + WMMA: both A and W use {loop-invariant uniform
// base + 32-bit per-lane offset + instruction immediate}; offsets advance
// with one v_add_nc_u32 each per k-step. No 64-bit VGPR addresses ->
// no WMMA WAR hazard NOPs, near-zero per-iteration VALU.
// ---------------------------------------------------------------------------

typedef __attribute__((ext_vector_type(16))) __bf16 v16bf;
typedef __attribute__((ext_vector_type(8)))  __bf16 v8bf;
typedef __attribute__((ext_vector_type(8)))  float  v8f;

#define B_     1024
#define N_     32
#define E_     31
#define E2_    62
#define NTOT_  (B_*N_)    // 32768 node rows
#define METOT_ (B_*E2_)   // 63488 edge rows

union Frag16 { v16bf v; v8bf h[2]; };

__device__ __forceinline__ v8f wmma_bf16(v16bf a, v16bf b, v8f c) {
  return __builtin_amdgcn_wmma_f32_16x16x32_bf16(
      false, a, false, b, (short)0, c, false, false);
}

__device__ __forceinline__ void split_store(float v, __bf16* hi, __bf16* lo,
                                            size_t idx) {
  __bf16 h = (__bf16)v;
  hi[idx] = h;
  lo[idx] = (__bf16)(v - (float)h);
}

__device__ __forceinline__ float nodefeat(const float* __restrict__ state,
                                          const float* __restrict__ nattr,
                                          int v, int k) {
  return (k < 64) ? state[(size_t)v * 64 + k] : nattr[(v & 31) * 16 + (k - 64)];
}

// ---------------------------------------------------------------------------
// Weight packing, kt-major: element ((kt*ntt + nt)*32 + lane)*16 + j holds
// W[kt*32 + (lane<16?0:8) + j + (j<8?0:8)][nt*16 + (lane&15)]  (zero-padded past K).
// Per k-step a wave reads 4 fragments at consecutive 1024B immediates.
// ---------------------------------------------------------------------------
__global__ __launch_bounds__(256) void pack_w(const float* __restrict__ W,
                                              int K, int Kp, int Nc,
                                              __bf16* __restrict__ hi,
                                              __bf16* __restrict__ lo) {
  int t = blockIdx.x * 256 + threadIdx.x;     // frag*32 + lane
  int ktn = Kp >> 5;
  int ntt = Nc >> 4;
  if (t >= ktn * ntt * 32) return;
  int frag = t >> 5, lane = t & 31;
  int kt = frag / ntt, nt = frag - kt * ntt;
  int n = nt * 16 + (lane & 15);
  int kbase = kt * 32 + ((lane < 16) ? 0 : 8);
  int base = t * 16;
#pragma unroll
  for (int j = 0; j < 16; ++j) {
    int k = kbase + j + ((j < 8) ? 0 : 8);
    float v = (k < K) ? W[(size_t)k * Nc + n] : 0.0f;
    split_store(v, hi, lo, base + j);
  }
}

// ---------------------------------------------------------------------------
// K-loop core over one contiguous A section. W offset (32-bit, bytes)
// advances by wstep_b per k-step and is threaded across sections by ref.
// ---------------------------------------------------------------------------
__device__ __forceinline__ void kloop(
    int nk,
    const char* __restrict__ abh, const char* __restrict__ abl,
    uint32_t aoff,                          // per-lane byte offset in section
    const char* __restrict__ wbh, const char* __restrict__ wbl,
    uint32_t& woff, uint32_t wstep_b, v8f (&acc)[4]) {
  for (int kk = 0; kk < nk; ++kk) {
    Frag16 ah, al;
    ah.h[0] = *(const v8bf*)(abh + aoff);
    ah.h[1] = *(const v8bf*)(abh + aoff + 32);
    al.h[0] = *(const v8bf*)(abl + aoff);
    al.h[1] = *(const v8bf*)(abl + aoff + 32);
#pragma unroll
    for (int t = 0; t < 4; ++t) {
      v16bf bh = *(const v16bf*)(wbh + woff + t * 1024);
      v16bf bl = *(const v16bf*)(wbl + woff + t * 1024);
      acc[t] = wmma_bf16(ah.v, bh, acc[t]);   // hi*hi
      acc[t] = wmma_bf16(ah.v, bl, acc[t]);   // hi*lo
      acc[t] = wmma_bf16(al.v, bh, acc[t]);   // lo*hi
    }
    aoff += 64;          // next 32-elem K block (bytes)
    woff += wstep_b;     // next kt chunk of packed W (bytes)
  }
}

// ---------------------------------------------------------------------------
// GEMM: C[M x Nc] = act(A[M x Kp] @ W + bias).
// Block = 8 waves = 32(M) x 256(N); wave owns 16 x 64 (4 n-tiles).
// MODE 0: dense planes (lda). MODE 1: [P0[idx0]|P0[idx1]|P2[r]] 256|256|256.
// MODE 2: [P0[r]|P2[r]] 256|256.  EPI: 1=relu, 2=store hi/lo, 4=store f32.
// ---------------------------------------------------------------------------
template <int MODE, int EPI>
__global__ __launch_bounds__(256) void gemm_bf16x3(
    const __bf16* __restrict__ Ahi, const __bf16* __restrict__ Alo, int lda,
    const __bf16* __restrict__ P0hi, const __bf16* __restrict__ P0lo,
    const __bf16* __restrict__ P2hi, const __bf16* __restrict__ P2lo,
    const int* __restrict__ idx0, const int* __restrict__ idx1,
    const __bf16* __restrict__ Whi, const __bf16* __restrict__ Wlo,
    const float* __restrict__ bias,
    float* __restrict__ Cf, __bf16* __restrict__ Chi, __bf16* __restrict__ Clo,
    int Kp, int Nc) {
  const int lane = threadIdx.x & 31;
  const int wave = threadIdx.x >> 5;
  // Force wave-uniform scalars: scalar branch only, no EXEC-masked WMMA.
  const int mtu = __builtin_amdgcn_readfirstlane(wave >> 2);
  const int nqu = __builtin_amdgcn_readfirstlane(wave & 3);
  if (nqu * 64 >= Nc) return;

  const int lrow = lane & 15;
  const int row  = blockIdx.x * 32 + mtu * 16 + lrow;
  const int kAe  = (lane < 16) ? 0 : 8;       // A-fragment K offset (elems)
  const uint32_t wstep_b = (uint32_t)(Nc >> 4) * 1024;  // packed-W kt stride

  // Loop-invariant uniform W bases (per wave) + 32-bit running voffset.
  const char* wbh = (const char*)(Whi + nqu * 2048);
  const char* wbl = (const char*)(Wlo + nqu * 2048);
  uint32_t woff = (uint32_t)lane * 32;

  v8f acc[4] = {};

  if (MODE == 0) {
    uint32_t aoff = (uint32_t)(row * lda + kAe) * 2;
    __builtin_prefetch((const char*)Ahi + aoff, 0, 1);  // global_prefetch_b8
    kloop(Kp >> 5, (const char*)Ahi, (const char*)Alo, aoff,
          wbh, wbl, woff, wstep_b, acc);
  } else if (MODE == 1) {
    const int ri0 = idx0[row], ri1 = idx1[row];
    kloop(8, (const char*)P0hi, (const char*)P0lo,
          (uint32_t)(ri0 * 256 + kAe) * 2, wbh, wbl, woff, wstep_b, acc);
    kloop(8, (const char*)P0hi, (const char*)P0lo,
          (uint32_t)(ri1 * 256 + kAe) * 2, wbh, wbl, woff, wstep_b, acc);
    kloop(8, (const char*)P2hi, (const char*)P2lo,
          (uint32_t)(row * 256 + kAe) * 2, wbh, wbl, woff, wstep_b, acc);
  } else {
    kloop(8, (const char*)P0hi, (const char*)P0lo,
          (uint32_t)(row * 256 + kAe) * 2, wbh, wbl, woff, wstep_b, acc);
    kloop(8, (const char*)P2hi, (const char*)P2lo,
          (uint32_t)(row * 256 + kAe) * 2, wbh, wbl, woff, wstep_b, acc);
  }

  // C/D layout: VGPR j, lane L -> M = j + (L<16?0:8), N = L&15
  const int rbase = blockIdx.x * 32 + mtu * 16 + ((lane < 16) ? 0 : 8);
#pragma unroll
  for (int t = 0; t < 4; ++t) {
    const int nc = (nqu * 4 + t) * 16 + lrow;
    const float bv = bias[nc];
#pragma unroll
    for (int j = 0; j < 8; ++j) {
      float v = acc[t][j] + bv;
      if (EPI & 1) v = fmaxf(v, 0.0f);
      size_t o = (size_t)(rbase + j) * Nc + nc;
      if (EPI & 4) Cf[o] = v;
      if (EPI & 2) split_store(v, Chi, Clo, o);
    }
  }
}

// ---------------------------------------------------------------------------
// Gather / concat / scatter helpers (emit hi/lo planes)
// ---------------------------------------------------------------------------
__global__ __launch_bounds__(256) void build_einp1(
    const float* __restrict__ state, const float* __restrict__ action,
    const float* __restrict__ nattr, const float* __restrict__ eattr,
    const int* __restrict__ rol, const int* __restrict__ col,
    __bf16* __restrict__ Ahi, __bf16* __restrict__ Alo) {
  const int KP = 192;                      // 185 padded to 192
  int idx = blockIdx.x * 256 + threadIdx.x;
  if (idx >= METOT_ * KP) return;
  int r = idx / KP, k = idx - r * KP;
  int e = r % E2_;
  int b = r / E2_;
  float v;
  if (k < 80)       v = nodefeat(state, nattr, rol[r], k);
  else if (k < 160) v = nodefeat(state, nattr, col[r], k - 80);
  else if (k < 168) v = action[((size_t)b * E_ + (e % E_)) * 8 + (k - 160)];
  else if (k < 185) v = eattr[e * 17 + (k - 168)];
  else              v = 0.0f;
  split_store(v, Ahi, Alo, idx);
}

__global__ __launch_bounds__(256) void build_ninp1(
    const float* __restrict__ state, const float* __restrict__ nattr,
    const float* __restrict__ agg,
    __bf16* __restrict__ Ahi, __bf16* __restrict__ Alo) {
  const int KP = 352;                      // 336 padded to 352
  int idx = blockIdx.x * 256 + threadIdx.x;
  if (idx >= NTOT_ * KP) return;
  int r = idx / KP, k = idx - r * KP;
  float v;
  if (k < 80)       v = nodefeat(state, nattr, r, k);
  else if (k < 336) v = agg[(size_t)r * 256 + (k - 80)];
  else              v = 0.0f;
  split_store(v, Ahi, Alo, idx);
}

__global__ __launch_bounds__(256) void zero_f32(float* __restrict__ p, int n) {
  int i = blockIdx.x * 256 + threadIdx.x;
  if (i < n) p[i] = 0.0f;
}

__global__ __launch_bounds__(256) void cvt_hl(const float* __restrict__ src,
                                              __bf16* __restrict__ hi,
                                              __bf16* __restrict__ lo, int n) {
  int i = blockIdx.x * 256 + threadIdx.x;
  if (i < n) split_store(src[i], hi, lo, i);
}

__global__ __launch_bounds__(256) void scatter_add_edges(
    const __bf16* __restrict__ Ehi, const __bf16* __restrict__ Elo,
    const int* __restrict__ rol, float* __restrict__ agg) {
  int idx = blockIdx.x * 256 + threadIdx.x;   // exactly METOT_*256 threads
  int r = idx >> 8, c = idx & 255;
  float v = (float)Ehi[idx] + (float)Elo[idx];
  __hip_atomic_fetch_add(&agg[(size_t)rol[r] * 256 + c], v,
                         __ATOMIC_RELAXED, __HIP_MEMORY_SCOPE_AGENT);
}

// ---------------------------------------------------------------------------
// Host orchestration
// ---------------------------------------------------------------------------
struct LayerDef { int K, Kp, Nc, win; size_t off; };

extern "C" void kernel_launch(void* const* d_in, const int* in_sizes, int n_in,
                              void* d_out, int out_size, void* d_ws, size_t ws_size,
                              hipStream_t stream) {
  (void)in_sizes; (void)n_in; (void)out_size; (void)ws_size;

  const float* state  = (const float*)d_in[0];
  const float* action = (const float*)d_in[1];
  const float* nattr  = (const float*)d_in[2];
  const float* eattr  = (const float*)d_in[3];
  const int*   rol    = (const int*)d_in[28];
  const int*   col    = (const int*)d_in[29];
  float* out = (float*)d_out;

  static const LayerDef L[12] = {
    {185, 192, 256,  4,      0},   // p1e.W1
    {256, 256, 256,  6,  49152},   // p1e.W2
    {256, 256, 256,  8, 114688},   // p1e.W3
    {336, 352, 256, 10, 180224},   // p1n.W1
    {256, 256, 256, 12, 270336},   // p1n.W2
    {256, 256, 256, 14, 335872},   // p1n.W3
    {768, 768, 256, 16, 401408},   // p2e.W1
    {256, 256, 256, 18, 598016},   // p2e.W2
    {256, 256, 256, 20, 663552},   // p2e.W3
    {512, 512, 256, 22, 729088},   // p2n.W1
    {256, 256, 256, 24, 860160},   // p2n.W2
    {256, 256,  64, 26, 925696},   // p2n.W3
  };
  const size_t WELEMS = 942080;

  char* p = (char*)d_ws;
  auto carve = [&](size_t bytes) -> void* {
    void* q = (void*)p; p += (bytes + 255) & ~(size_t)255; return q;
  };
  __bf16* Whi = (__bf16*)carve(WELEMS * 2);
  __bf16* Wlo = (__bf16*)carve(WELEMS * 2);
  __bf16* A1h = (__bf16*)carve((size_t)METOT_ * 192 * 2);  // einp1 / ninp1
  __bf16* A1l = (__bf16*)carve((size_t)METOT_ * 192 * 2);
  __bf16* H1h = (__bf16*)carve((size_t)METOT_ * 256 * 2);
  __bf16* H1l = (__bf16*)carve((size_t)METOT_ * 256 * 2);
  __bf16* H2h = (__bf16*)carve((size_t)METOT_ * 256 * 2);
  __bf16* H2l = (__bf16*)carve((size_t)METOT_ * 256 * 2);
  __bf16* EDh = (__bf16*)carve((size_t)METOT_ * 256 * 2);  // edge1, later edge2
  __bf16* EDl = (__bf16*)carve((size_t)METOT_ * 256 * 2);
  __bf16* N1h = (__bf16*)carve((size_t)NTOT_ * 256 * 2);   // node1
  __bf16* N1l = (__bf16*)carve((size_t)NTOT_ * 256 * 2);
  __bf16* AGh = (__bf16*)carve((size_t)NTOT_ * 256 * 2);   // agg hi/lo
  __bf16* AGl = (__bf16*)carve((size_t)NTOT_ * 256 * 2);
  float*  AGG = (float*)carve((size_t)NTOT_ * 256 * 4);    // agg f32 accum

  for (int i = 0; i < 12; ++i) {
    int threads = (L[i].Kp / 32) * (L[i].Nc / 16) * 32;
    pack_w<<<(threads + 255) / 256, 256, 0, stream>>>(
        (const float*)d_in[L[i].win], L[i].K, L[i].Kp, L[i].Nc,
        Whi + L[i].off, Wlo + L[i].off);
  }

  const int GE = METOT_ / 32, GN = NTOT_ / 32;

  // EPI: 1=relu, 2=store hi/lo, 4=store f32
  auto dense = [&](auto epi_tag, const __bf16* ah, const __bf16* al, int li,
                   float* cf, __bf16* ch, __bf16* cl, int M) {
    constexpr int EPI = decltype(epi_tag)::value;
    gemm_bf16x3<0, EPI><<<dim3(M / 32), 256, 0, stream>>>(
        ah, al, L[li].Kp, nullptr, nullptr, nullptr, nullptr, nullptr, nullptr,
        Whi + L[li].off, Wlo + L[li].off, (const float*)d_in[L[li].win + 1],
        cf, ch, cl, L[li].Kp, L[li].Nc);
  };
  using E3 = std::integral_constant<int, 3>;   // relu + hi/lo
  using E2t = std::integral_constant<int, 2>;  // linear + hi/lo
  using E4 = std::integral_constant<int, 4>;   // linear + f32

  // ---- block 1: edge MLP ----
  build_einp1<<<(METOT_ * 192) / 256, 256, 0, stream>>>(
      state, action, nattr, eattr, rol, col, A1h, A1l);
  dense(E3{},  A1h, A1l, 0, nullptr, H1h, H1l, METOT_);
  dense(E3{},  H1h, H1l, 1, nullptr, H2h, H2l, METOT_);
  dense(E2t{}, H2h, H2l, 2, nullptr, EDh, EDl, METOT_);     // edge1

  zero_f32<<<(NTOT_ * 256) / 256, 256, 0, stream>>>(AGG, NTOT_ * 256);
  scatter_add_edges<<<(METOT_ * 256) / 256, 256, 0, stream>>>(EDh, EDl, rol, AGG);

  // ---- block 1: node MLP ----
  build_ninp1<<<(NTOT_ * 352) / 256, 256, 0, stream>>>(state, nattr, AGG, A1h, A1l);
  dense(E3{},  A1h, A1l, 3, nullptr, H1h, H1l, NTOT_);
  dense(E3{},  H1h, H1l, 4, nullptr, H2h, H2l, NTOT_);
  dense(E2t{}, H2h, H2l, 5, nullptr, N1h, N1l, NTOT_);      // node1

  // ---- block 2: edge MLP (fused concat: node1[rol] | node1[col] | edge1) ----
  gemm_bf16x3<1, 3><<<dim3(GE), 256, 0, stream>>>(
      nullptr, nullptr, 0, N1h, N1l, EDh, EDl, rol, col,
      Whi + L[6].off, Wlo + L[6].off, (const float*)d_in[17],
      nullptr, H1h, H1l, 768, 256);
  dense(E3{},  H1h, H1l, 7, nullptr, H2h, H2l, METOT_);
  dense(E2t{}, H2h, H2l, 8, nullptr, EDh, EDl, METOT_);     // edge2

  zero_f32<<<(NTOT_ * 256) / 256, 256, 0, stream>>>(AGG, NTOT_ * 256);
  scatter_add_edges<<<(METOT_ * 256) / 256, 256, 0, stream>>>(EDh, EDl, rol, AGG);
  cvt_hl<<<(NTOT_ * 256) / 256, 256, 0, stream>>>(AGG, AGh, AGl, NTOT_ * 256);

  // ---- block 2: node MLP (fused concat: node1 | agg2) ----
  gemm_bf16x3<2, 3><<<dim3(GN), 256, 0, stream>>>(
      nullptr, nullptr, 0, N1h, N1l, AGh, AGl, nullptr, nullptr,
      Whi + L[9].off, Wlo + L[9].off, (const float*)d_in[23],
      nullptr, H1h, H1l, 512, 256);
  dense(E3{},  H1h, H1l, 10, nullptr, H2h, H2l, NTOT_);
  dense(E4{},  H2h, H2l, 11, out, nullptr, nullptr, NTOT_); // 32768 x 64 out
}
